// FundLSTM_6107443495000
// MI455X (gfx1250) — matrix operand order repaired and checked by reading, not compiled, not run on hardware
//
#include <hip/hip_runtime.h>
#include <math.h>

typedef _Float16 v16h __attribute__((ext_vector_type(16)));
typedef _Float16 v8h  __attribute__((ext_vector_type(8)));
typedef float    v8f  __attribute__((ext_vector_type(8)));

#define Bsz  1024
#define Tlen 128
#define Hdim 28
#define Gdim 112
#define NT   7      // 112 / 16 gate tiles

// Pack two f16 into a dword.
__device__ __forceinline__ unsigned pack2(_Float16 a, _Float16 b) {
    union { _Float16 h[2]; unsigned u; } p;
    p.h[0] = a; p.h[1] = b;
    return p.u;
}

// One wave32 per 16-row batch tile; runs one LSTM layer over the full sequence.
// KSTEPS==2 : layer 0 (F=38, f32 input). KSTEPS==1 : layers 1..15 (F=28, f16 input).
// All weight B-tiles live in VGPRs (occupancy is irrelevant: only B/16=64 waves exist).
template <int KSTEPS>
__global__ __launch_bounds__(32) void lstm_layer(
    const void*  __restrict__ xin_,  // [B,T,F] f32 (layer0) or f16 (h_seq)
    const float* __restrict__ Wih,   // [G, F]
    const float* __restrict__ Whh,   // [G, H]
    const float* __restrict__ bias,  // [G]
    _Float16*    __restrict__ hout)  // [B, T, H] f16
{
    constexpr int F = (KSTEPS == 2) ? 38 : 28;
    const int lane = threadIdx.x;
    const int lo   = lane & 15;   // A row / B,D column
    const int hi   = lane >> 4;   // half-wave selector
    const int b0   = blockIdx.x * 16;

    // f16 activation tiles, zero-padded so A-gathers need no bounds checks.
    __shared__ _Float16 sX16[16][64];   // input rows for current t (cols >= F stay 0)
    __shared__ _Float16 sH16[16][32];   // h_{t-1} (cols >= Hdim stay 0)
    __shared__ float    sG[16][Gdim];   // gate pre-activations

    // ---- Weights -> registers, pre-laid-out as WMMA B tiles (32x16 f16), contiguous k order (matches A):
    // lane -> column N = lane%16 ; K = 16*(lane>=16) + e.
    v16h WBx[NT * KSTEPS];
#pragma unroll
    for (int n = 0; n < NT; ++n)
#pragma unroll
        for (int ks = 0; ks < KSTEPS; ++ks) {
            const int g = n * 16 + lo;
            v16h w;
#pragma unroll
            for (int e = 0; e < 16; ++e) {
                const int K = ks * 32 + hi * 16 + e;
                w[e] = (K < F) ? (_Float16)Wih[g * F + K] : (_Float16)0.0f;
            }
            WBx[n * KSTEPS + ks] = w;
        }
    v16h WBh[NT];
#pragma unroll
    for (int n = 0; n < NT; ++n) {
        const int g = n * 16 + lo;
        v16h w;
#pragma unroll
        for (int e = 0; e < 16; ++e) {
            const int K = hi * 16 + e;
            w[e] = (K < Hdim) ? (_Float16)Whh[g * Hdim + K] : (_Float16)0.0f;
        }
        WBh[n] = w;
    }

    // Bias per N column, hoisted out of the time loop.
    float bj[NT];
#pragma unroll
    for (int n = 0; n < NT; ++n) bj[n] = bias[n * 16 + lo];

    // Zero LDS tiles once (provides h0 = 0 and permanent zero padding).
    for (int w = lane; w < 16 * 64 / 2; w += 32) ((unsigned*)sX16)[w] = 0u;
    for (int w = lane; w < 16 * 32 / 2; w += 32) ((unsigned*)sH16)[w] = 0u;

    // Cell state: lane owns (row lo, units hi*14 .. hi*14+13).
    float creg[14];
#pragma unroll
    for (int i = 0; i < 14; ++i) creg[i] = 0.0f;
    __syncthreads();

    for (int t = 0; t < Tlen; ++t) {
        // ---- Stage input rows for timestep t into sX16 (f16) ----
        if constexpr (KSTEPS == 2) {
            const float* xin = (const float*)xin_;
            // 16 rows x 19 dword-pairs (cols 0..37), all in-bounds: branch-free.
            for (int w = lane; w < 16 * 19; w += 32) {
                const int row = w / 19, pr = w % 19;
                const float* src = xin + ((size_t)(b0 + row) * Tlen + t) * F + 2 * pr;
                ((unsigned*)&sX16[row][0])[pr] = pack2((_Float16)src[0], (_Float16)src[1]);
            }
            if (t + 1 < Tlen)
                __builtin_prefetch(&xin[((size_t)(b0 + lo) * Tlen + (t + 1)) * F], 0, 1);
        } else {
            const _Float16* xin = (const _Float16*)xin_;
            // Already f16: straight dword copy, 16 rows x 14 dwords.
            for (int w = lane; w < 16 * 14; w += 32) {
                const int row = w / 14, pr = w % 14;
                const unsigned* src =
                    (const unsigned*)(xin + ((size_t)(b0 + row) * Tlen + t) * Hdim);
                ((unsigned*)&sX16[row][0])[pr] = src[pr];
            }
            if (t + 1 < Tlen)
                __builtin_prefetch(&xin[((size_t)(b0 + lo) * Tlen + (t + 1)) * Hdim], 0, 1);
        }
        __syncthreads();

        // ---- A-operand gathers: two contiguous 16B chunks per lane, no cvt ----
        // A 16x32 f16, contiguous k order (same as B): row M = lane%16, K = 16*(lane>=16) + e.
        v16h Ax[KSTEPS];
#pragma unroll
        for (int ks = 0; ks < KSTEPS; ++ks) {
            const v8h c0 = *(const v8h*)&sX16[lo][ks * 32 + hi * 16];
            const v8h c1 = *(const v8h*)&sX16[lo][ks * 32 + hi * 16 + 8];
            Ax[ks] = __builtin_shufflevector(c0, c1, 0, 1, 2, 3, 4, 5, 6, 7,
                                             8, 9, 10, 11, 12, 13, 14, 15);
        }
        const v8h h0 = *(const v8h*)&sH16[lo][hi * 16];
        const v8h h1 = *(const v8h*)&sH16[lo][hi * 16 + 8];
        const v16h Ah = __builtin_shufflevector(h0, h1, 0, 1, 2, 3, 4, 5, 6, 7,
                                                8, 9, 10, 11, 12, 13, 14, 15);

        // ---- Gate GEMM: 7 N-tiles, acc seeded with bias (N = lane%16) ----
#pragma unroll
        for (int n = 0; n < NT; ++n) {
            const float b = bj[n];
            v8f acc = {b, b, b, b, b, b, b, b};
            acc = __builtin_amdgcn_wmma_f32_16x16x32_f16(
                false, Ax[0], false, WBx[n * KSTEPS + 0], (short)0, acc, false, false);
            if constexpr (KSTEPS == 2)
                acc = __builtin_amdgcn_wmma_f32_16x16x32_f16(
                    false, Ax[1], false, WBx[n * KSTEPS + 1], (short)0, acc, false, false);
            acc = __builtin_amdgcn_wmma_f32_16x16x32_f16(
                false, Ah, false, WBh[n], (short)0, acc, false, false);
            // D layout: M = v + 8*(lane>=16), N = lane%16
#pragma unroll
            for (int v = 0; v < 8; ++v)
                sG[v + 8 * hi][n * 16 + lo] = acc[v];
        }
        __syncthreads();

        // ---- Elementwise LSTM cell ----
        // Lane owns row r = lo, units u = hi*14 + i : contiguous gate reads,
        // packed dword h stores to LDS and global.
        {
            const int r  = lo;
            const int ub = hi * 14;
            float hv[14];
#pragma unroll
            for (int i = 0; i < 14; ++i) {
                const int u = ub + i;
                const float ig = sG[r][u];
                const float fg = sG[r][Hdim + u];
                const float gg = sG[r][2 * Hdim + u];
                const float og = sG[r][3 * Hdim + u];
                const float si = 1.0f / (1.0f + expf(-ig));
                const float sf = 1.0f / (1.0f + expf(-fg));
                const float so = 1.0f / (1.0f + expf(-og));
                const float c  = sf * creg[i] + si * tanhf(gg);
                creg[i] = c;
                hv[i]   = so * tanhf(c);
            }
            __syncthreads();   // gate reads done before sH16 overwrite below
            unsigned* ldsdst = (unsigned*)&sH16[r][ub];
            unsigned* gdst =
                (unsigned*)(hout + ((size_t)(b0 + r) * Tlen + t) * Hdim + ub);
#pragma unroll
            for (int j = 0; j < 7; ++j) {
                const unsigned p = pack2((_Float16)hv[2 * j], (_Float16)hv[2 * j + 1]);
                ldsdst[j] = p;
                gdst[j]   = p;
            }
        }
        __syncthreads();
    }
}

// Linear head on the last 61 timesteps: out[b, tt] = h[b, T-61+tt, :] . Wout + bout
__global__ __launch_bounds__(256) void head_kernel(
    const _Float16* __restrict__ hseq, const float* __restrict__ Wout,
    const float* __restrict__ bout, float* __restrict__ out)
{
    const int idx = blockIdx.x * blockDim.x + threadIdx.x;
    if (idx >= Bsz * 61) return;
    const int b  = idx / 61;
    const int tt = idx % 61;
    const int t  = Tlen - 61 + tt;
    const _Float16* h = &hseq[((size_t)b * Tlen + t) * Hdim];
    float s = bout[0];
#pragma unroll
    for (int j = 0; j < Hdim; ++j) s += (float)h[j] * Wout[j];
    out[idx] = s;
}

extern "C" void kernel_launch(void* const* d_in, const int* in_sizes, int n_in,
                              void* d_out, int out_size, void* d_ws, size_t ws_size,
                              hipStream_t stream) {
    const float* x    = (const float*)d_in[0];   // [B, T, 38]
    const float* Wih0 = (const float*)d_in[1];   // [112, 38]
    const float* Wihr = (const float*)d_in[2];   // [15, 112, 28]
    const float* Whh  = (const float*)d_in[3];   // [16, 112, 28]
    const float* bias = (const float*)d_in[4];   // [16, 112]
    const float* Wout = (const float*)d_in[5];   // [1, 28]
    const float* bout = (const float*)d_in[6];   // [1]
    float* out = (float*)d_out;

    const size_t hsz = (size_t)Bsz * Tlen * Hdim;          // elements
    _Float16* hbuf0 = (_Float16*)d_ws;
    _Float16* hbuf1 = hbuf0 + hsz;
    _Float16* hbuf[2] = {hbuf0, hbuf1};

    // Layer 0: f32 input, F=38, two K-steps.
    lstm_layer<2><<<Bsz / 16, 32, 0, stream>>>(
        x, Wih0, Whh, bias, hbuf[0]);
    // Layers 1..15: f16 input (previous h_seq), F=28, one K-step.
    for (int l = 1; l < 16; ++l) {
        lstm_layer<1><<<Bsz / 16, 32, 0, stream>>>(
            hbuf[(l - 1) & 1],
            Wihr + (size_t)(l - 1) * Gdim * Hdim,
            Whh + (size_t)l * Gdim * Hdim,
            bias + (size_t)l * Gdim,
            hbuf[l & 1]);
    }
    // Layer 15 wrote hbuf[15 & 1] = hbuf1
    head_kernel<<<(Bsz * 61 + 255) / 256, 256, 0, stream>>>(hbuf1, Wout, bout, out);
}
